// ViT_91130616087325
// MI455X (gfx1250) — matrix operand order repaired and checked by reading, not compiled, not run on hardware
//
#include <hip/hip_runtime.h>
#include <hip/hip_bf16.h>
#include <math.h>

typedef __attribute__((ext_vector_type(16))) _Float16 v16h;
typedef __attribute__((ext_vector_type(8)))  _Float16 v8h;
typedef __attribute__((ext_vector_type(8)))  float    v8f;
typedef __attribute__((ext_vector_type(4)))  unsigned int v4u;
typedef __attribute__((ext_vector_type(8)))  int v8i;
typedef __attribute__((ext_vector_type(4)))  int v4i;

#if __has_builtin(__builtin_amdgcn_tensor_load_to_lds) && \
    __has_builtin(__builtin_amdgcn_s_wait_tensorcnt)
#define HAVE_TDM 1
#endif

// ---------------- model constants ----------------
#define BVIT   32
#define SEQ    197
#define SEQP   224      // padded to 7 * 32 for WMMA K-steps
#define DIM    768
#define NHEAD  12
#define DHEAD  64
#define NLAYER 12
#define FFDIM  3072
#define QKVLD  2304
#define NPATCH 196
#define NCLS   1000
#define QTILES 13       // ceil(197/16)

#define ROWS   (BVIT * SEQ)        // 6304
#define EROWS  (BVIT * NPATCH)     // 6272

#define SHUF16(lo, hi) \
  __builtin_shufflevector(lo, hi, 0, 1, 2, 3, 4, 5, 6, 7, 8, 9, 10, 11, 12, 13, 14, 15)

#ifdef HAVE_TDM
// 2D f16 tile DMA: global (row-major, stride elems) -> LDS (contiguous).
// Tensor dims are given relative to the tile origin so TDM zero-fills OOB rows.
__device__ __forceinline__ void tdm_load_tile_f16(
    unsigned lds_off, const _Float16* gsrc, unsigned tile_d0, unsigned tile_d1,
    unsigned tens_d0, unsigned tens_d1, unsigned long long stride) {
  const unsigned long long ga = (unsigned long long)(size_t)gsrc;
  v4u g0;
  g0[0] = 1u;                                            // count=1, user D#
  g0[1] = lds_off;                                       // lds_addr (bytes)
  g0[2] = (unsigned)ga;                                  // global_addr[31:0]
  g0[3] = (unsigned)((ga >> 32) & 0x01ffffffu) | 0x80000000u;  // addr[56:32] | type=2
  v8i g1;
  g1[0] = (int)(1u << 16);                               // data_size=1 (2 bytes)
  g1[1] = (int)((tens_d0 & 0xffffu) << 16);              // tensor_dim0[15:0]
  g1[2] = (int)(((tens_d0 >> 16) & 0xffffu) | ((tens_d1 & 0xffffu) << 16));
  g1[3] = (int)(((tens_d1 >> 16) & 0xffffu) | ((tile_d0 & 0xffffu) << 16));
  g1[4] = (int)(tile_d1 & 0xffffu);                      // tile_dim1 (tile_dim2=0)
  g1[5] = (int)(unsigned)stride;                         // dim0_stride[31:0]
  g1[6] = (int)(unsigned)((stride >> 32) & 0xffffu);     // dim0_stride[47:32]
  g1[7] = 0;
  v4i gz4 = {0, 0, 0, 0};
  v8i gz8 = {0, 0, 0, 0, 0, 0, 0, 0};
  __builtin_amdgcn_tensor_load_to_lds(g0, g1, gz4, gz4, gz8, 0);
}
#endif

// ---------------- generic WMMA GEMM ----------------
// Out[M,N] = act(A16[M,K] @ Bw[K,N] + bias) (+ Res)
// A is f16 row-major; Bw/bias/Res are f32; OUT16 selects f16/f32 output.
#define BM 128
#define BN 128
#define BK 32

template <int ACT, int OUT16>
__global__ __launch_bounds__(256) void gemm_wmma(
    const _Float16* __restrict__ A, const float* __restrict__ Bw,
    const float* __restrict__ bias, const float* __restrict__ Res,
    void* __restrict__ OutP, int M, int N, int K) {
  __shared__ _Float16 As[2][BM][BK];   // [m][k]
  __shared__ _Float16 Bs[2][BN][BK];   // transposed: [n][k]

  const int tid  = threadIdx.x;
  const int lane = tid & 31;
  const int wave = tid >> 5;
  const int bm = blockIdx.y * BM;
  const int bn = blockIdx.x * BN;
  const int wr = (wave & 3) * 32;
  const int wc = (wave >> 2) * 64;
  const int h2 = lane >> 4;
  const int l16 = lane & 15;

  v8f acc[2][4];
#pragma unroll
  for (int i = 0; i < 2; ++i)
#pragma unroll
    for (int j = 0; j < 4; ++j) acc[i][j] = (v8f){0, 0, 0, 0, 0, 0, 0, 0};

  const int nk = K / BK;

  auto stageA = [&](int buf, int k0) {
#ifdef HAVE_TDM
    if (wave == 0) {
      const unsigned lds_off = (unsigned)(size_t)&As[buf][0][0];
      tdm_load_tile_f16(lds_off, A + (size_t)bm * K + k0, BK, BM,
                        (unsigned)(K - k0), (unsigned)(M - bm),
                        (unsigned long long)K);
    }
#else
    const int ar = tid >> 1;
    const int ac = (tid & 1) * 16;
    const int gr = bm + ar;
    v8h a0 = (v8h)(_Float16)0.f, a1 = (v8h)(_Float16)0.f;
    if (gr < M) {
      const _Float16* src = A + (size_t)gr * K + k0 + ac;
      a0 = ((const v8h*)src)[0];
      a1 = ((const v8h*)src)[1];
    }
    *(v8h*)&As[buf][ar][ac]     = a0;
    *(v8h*)&As[buf][ar][ac + 8] = a1;
#endif
  };

  auto stageB = [&](int buf, int k0) {
    const int bk  = tid >> 3;
    const int bn0 = (tid & 7) * 16;
    const float* bsrc = Bw + (size_t)(k0 + bk) * N + bn + bn0;
    if (bn + bn0 + 16 <= N) {
      __builtin_prefetch(bsrc + (size_t)BK * N, 0, 1);  // global_prefetch_b8
#pragma unroll
      for (int i = 0; i < 4; ++i) {
        float4 q = ((const float4*)bsrc)[i];
        Bs[buf][bn0 + i * 4 + 0][bk] = (_Float16)q.x;
        Bs[buf][bn0 + i * 4 + 1][bk] = (_Float16)q.y;
        Bs[buf][bn0 + i * 4 + 2][bk] = (_Float16)q.z;
        Bs[buf][bn0 + i * 4 + 3][bk] = (_Float16)q.w;
      }
    } else {
#pragma unroll
      for (int j = 0; j < 16; ++j) {
        float v = (bn + bn0 + j < N) ? bsrc[j] : 0.f;
        Bs[buf][bn0 + j][bk] = (_Float16)v;
      }
    }
  };

  auto compute = [&](int buf) {
    v16h afr[2];
    v16h bfr[4];
#pragma unroll
    for (int mt = 0; mt < 2; ++mt) {
      const int m = wr + mt * 16 + l16;
      v8h lo = *(const v8h*)&As[buf][m][h2 * 16];
      v8h hi = *(const v8h*)&As[buf][m][h2 * 16 + 8];
      afr[mt] = SHUF16(lo, hi);
    }
#pragma unroll
    for (int nt = 0; nt < 4; ++nt) {
      const int n = wc + nt * 16 + l16;
      bfr[nt] = *(const v16h*)&Bs[buf][n][h2 * 16];
    }
#pragma unroll
    for (int mt = 0; mt < 2; ++mt)
#pragma unroll
      for (int nt = 0; nt < 4; ++nt)
        acc[mt][nt] = __builtin_amdgcn_wmma_f32_16x16x32_f16(
            false, afr[mt], false, bfr[nt], (short)0, acc[mt][nt], false, false);
  };

  stageA(0, 0);
  stageB(0, 0);
#ifdef HAVE_TDM
  __builtin_amdgcn_s_wait_tensorcnt(0);
#endif
  __syncthreads();
  for (int s = 0; s < nk; ++s) {
    const int cur = s & 1;
    if (s + 1 < nk) {
      stageA(cur ^ 1, (s + 1) * BK);
      stageB(cur ^ 1, (s + 1) * BK);
    }
    compute(cur);
#ifdef HAVE_TDM
    __builtin_amdgcn_s_wait_tensorcnt(0);
#endif
    __syncthreads();
  }

  // ---- epilogue: bias, activation, residual, f32 or f16 store ----
#pragma unroll
  for (int mt = 0; mt < 2; ++mt) {
#pragma unroll
    for (int nt = 0; nt < 4; ++nt) {
      const int col = bn + wc + nt * 16 + l16;
      if (col >= N) continue;
      const float bcol = bias ? bias[col] : 0.f;
#pragma unroll
      for (int v = 0; v < 8; ++v) {
        const int row = bm + wr + mt * 16 + h2 * 8 + v;
        if (row < M) {
          float x = acc[mt][nt][v] + bcol;
          if (ACT == 1) x = x * 0.5f * (1.f + erff(x * 0.70710678118f));
          if (ACT == 2) x = fmaxf(x, 0.f);
          if (Res) x += Res[(size_t)row * N + col];
          if (OUT16)
            ((_Float16*)OutP)[(size_t)row * N + col] = (_Float16)x;
          else
            ((float*)OutP)[(size_t)row * N + col] = x;
        }
      }
    }
  }
}

// ---------------- WMMA attention (qkv is f16) ----------------
// One block = one (batch, head, 16-row query tile). 128 threads = 4 waves.
__global__ __launch_bounds__(128) void attn_wmma(
    const _Float16* __restrict__ qkv, _Float16* __restrict__ O) {
  __shared__ _Float16 Qs[16 * DHEAD];    //  2 KB
  __shared__ _Float16 KV[SEQP * DHEAD];  // 28 KB (K, then V^T)
  __shared__ float    Sc[16][SEQP];      // 14 KB
  __shared__ _Float16 Ph[16][SEQP];      //  7 KB
  __shared__ float    red[16][8];
  __shared__ float    rowmx[16], rowinv[16];

  const int id  = blockIdx.x;
  const int qt  = id % QTILES;
  const int hd  = (id / QTILES) % NHEAD;
  const int bb  = id / (QTILES * NHEAD);
  const int tid = threadIdx.x;
  const int lane = tid & 31;
  const int wave = tid >> 5;
  const int h2   = lane >> 4;
  const int l16  = lane & 15;
  const int q0   = qt * 16;
  const float scale = 0.125f;  // 1/sqrt(64)
  const size_t base = (size_t)bb * SEQ * QKVLD + hd * 192;

  // ---- stage Q (16x64) and K (SEQP x 64, zero padded), straight f16 copies
  {
    const int off = tid * 8;
    const int r = off >> 6, c = off & 63;
    const int qr = q0 + r;
    v8h hq = (v8h)(_Float16)0.f;
    if (qr < SEQ) hq = *(const v8h*)(qkv + base + (size_t)qr * QKVLD + c);
    *(v8h*)&Qs[r * 64 + c] = hq;
  }
#pragma unroll
  for (int i = 0; i < 14; ++i) {
    const int off = (i * 128 + tid) * 8;
    const int r = off >> 6, c = off & 63;
    v8h hk = (v8h)(_Float16)0.f;
    if (r < SEQ) hk = *(const v8h*)(qkv + base + 64 + (size_t)r * QKVLD + c);
    *(v8h*)&KV[r * 64 + c] = hk;
  }
  __syncthreads();

  // ---- scores: Q @ K^T ----
  v16h qf[2];
#pragma unroll
  for (int j = 0; j < 2; ++j) {
    v8h lo = *(const v8h*)&Qs[l16 * 64 + j * 32 + h2 * 16];
    v8h hi = *(const v8h*)&Qs[l16 * 64 + j * 32 + h2 * 16 + 8];
    qf[j] = SHUF16(lo, hi);
  }
  for (int kt = wave; kt < QTILES; kt += 4) {
    v8f sacc = (v8f){0, 0, 0, 0, 0, 0, 0, 0};
#pragma unroll
    for (int j = 0; j < 2; ++j) {
      v16h bf = *(const v16h*)&KV[(kt * 16 + l16) * 64 + j * 32 + h2 * 16];
      sacc = __builtin_amdgcn_wmma_f32_16x16x32_f16(false, qf[j], false, bf,
                                                    (short)0, sacc, false, false);
    }
#pragma unroll
    for (int v = 0; v < 8; ++v) Sc[h2 * 8 + v][kt * 16 + l16] = sacc[v] * scale;
  }
  __syncthreads();  // all KV reads + Sc writes done

  // ---- restage V transposed: KV[dh * SEQP + key] ----
#pragma unroll
  for (int i = 0; i < 14; ++i) {
    const int off = (i * 128 + tid) * 8;
    const int r = off >> 6, c = off & 63;
    v8h hv = (v8h)(_Float16)0.f;
    if (r < SEQ) hv = *(const v8h*)(qkv + base + 128 + (size_t)r * QKVLD + c);
#pragma unroll
    for (int j = 0; j < 8; ++j) KV[(c + j) * SEQP + r] = hv[j];
  }

  // ---- parallel softmax: row = tid&15, segment = tid>>4 (8 x 25 cols) ----
  const int srow = tid & 15, sseg = tid >> 4;
  const int cbeg = sseg * 25;
  const int cend = (cbeg + 25 < SEQ) ? cbeg + 25 : SEQ;
  float pm = -1e30f;
  for (int c = cbeg; c < cend; ++c) pm = fmaxf(pm, Sc[srow][c]);
  red[srow][sseg] = pm;
  __syncthreads();
  if (tid < 16) {
    float m = red[tid][0];
#pragma unroll
    for (int j = 1; j < 8; ++j) m = fmaxf(m, red[tid][j]);
    rowmx[tid] = m;
  }
  __syncthreads();
  const float mx = rowmx[srow];
  float ps = 0.f;
  for (int c = cbeg; c < cend; ++c) {
    float e = __expf(Sc[srow][c] - mx);
    Sc[srow][c] = e;
    ps += e;
  }
  red[srow][sseg] = ps;
  __syncthreads();
  if (tid < 16) {
    float s = 0.f;
#pragma unroll
    for (int j = 0; j < 8; ++j) s += red[tid][j];
    rowinv[tid] = 1.f / s;
  }
  __syncthreads();
  const float inv = rowinv[srow];
  for (int c = cbeg; c < cend; ++c) Ph[srow][c] = (_Float16)(Sc[srow][c] * inv);
  if (sseg == 7)
    for (int c = SEQ; c < SEQP; ++c) Ph[srow][c] = (_Float16)0.f;
  __syncthreads();

  // ---- O = P @ V ; wave w owns dh columns [w*16, w*16+16) ----
  const int c0 = wave * 16;
  v8f oacc = (v8f){0, 0, 0, 0, 0, 0, 0, 0};
#pragma unroll
  for (int step = 0; step < 7; ++step) {
    v8h plo = *(const v8h*)&Ph[l16][step * 32 + h2 * 16];
    v8h phi = *(const v8h*)&Ph[l16][step * 32 + h2 * 16 + 8];
    v16h af = SHUF16(plo, phi);
    v16h bf = *(const v16h*)&KV[(c0 + l16) * SEQP + step * 32 + h2 * 16];
    oacc = __builtin_amdgcn_wmma_f32_16x16x32_f16(false, af, false, bf,
                                                  (short)0, oacc, false, false);
  }
  const int col = hd * DHEAD + c0 + l16;
#pragma unroll
  for (int v = 0; v < 8; ++v) {
    const int qr = q0 + h2 * 8 + v;
    if (qr < SEQ) O[((size_t)bb * SEQ + qr) * DIM + col] = (_Float16)oacc[v];
  }
}

// ---------------- layernorm: f32 in, f16 out (one wave32 per row) -------
__global__ __launch_bounds__(256) void layernorm_rows(
    const float* __restrict__ X, const float* __restrict__ g,
    const float* __restrict__ bt, _Float16* __restrict__ Y, int M) {
  const int wave = threadIdx.x >> 5, lane = threadIdx.x & 31;
  const int row = blockIdx.x * 8 + wave;
  if (row >= M) return;
  const float* x = X + (size_t)row * DIM;
  float s = 0.f;
  for (int i = lane; i < DIM; i += 32) s += x[i];
#pragma unroll
  for (int o = 16; o > 0; o >>= 1) s += __shfl_xor(s, o, 32);
  const float mean = s * (1.f / DIM);
  float vv = 0.f;
  for (int i = lane; i < DIM; i += 32) {
    float d = x[i] - mean;
    vv += d * d;
  }
#pragma unroll
  for (int o = 16; o > 0; o >>= 1) vv += __shfl_xor(vv, o, 32);
  const float rstd = rsqrtf(vv * (1.f / DIM) + 1e-5f);
  for (int i = lane; i < DIM; i += 32)
    Y[(size_t)row * DIM + i] = (_Float16)((x[i] - mean) * rstd * g[i] + bt[i]);
}

// ---------------- patchify: f32 image -> f16 patch matrix ----------------
__global__ void patchify_k(const float* __restrict__ x, _Float16* __restrict__ Xp) {
  const size_t o = (size_t)blockIdx.x * blockDim.x + threadIdx.x;
  if (o >= (size_t)BVIT * NPATCH * DIM) return;
  const int b = o / (NPATCH * DIM);
  int flat = o % (NPATCH * DIM);  // C-major flat over (C, nh, nw, P, P)
  const int c  = flat / 50176;  flat %= 50176;   // 14*14*256
  const int ph = flat / 3584;   flat %= 3584;    // 14*256
  const int pw = flat / 256;    flat %= 256;
  const int py = flat / 16;
  const int px = flat % 16;
  Xp[o] = (_Float16)x[(((size_t)b * 3 + c) * 224 + (ph * 16 + py)) * 224 + (pw * 16 + px)];
}

// ---------------- assemble h = [cls ; patches] + pos (f32) ----------------
__global__ void assemble_k(const float* __restrict__ Xe, const float* __restrict__ cls,
                           const float* __restrict__ pos, float* __restrict__ h) {
  const size_t o = (size_t)blockIdx.x * blockDim.x + threadIdx.x;
  if (o >= (size_t)BVIT * SEQ * DIM) return;
  const int b = o / (SEQ * DIM);
  const int r = (o / DIM) % SEQ;
  const int d = o % DIM;
  float v = (r == 0) ? cls[d] : Xe[((size_t)b * NPATCH + (r - 1)) * DIM + d];
  h[o] = v + pos[(size_t)r * DIM + d];
}

__global__ void gather_cls_k(const float* __restrict__ h, _Float16* __restrict__ out) {
  const int o = blockIdx.x * blockDim.x + threadIdx.x;
  if (o >= BVIT * DIM) return;
  const int b = o / DIM, d = o % DIM;
  out[o] = (_Float16)h[(size_t)b * SEQ * DIM + d];
}

// ---------------- launcher ----------------
static inline dim3 ggrid(int M, int N) { return dim3((N + BN - 1) / BN, (M + BM - 1) / BM); }

extern "C" void kernel_launch(void* const* d_in, const int* in_sizes, int n_in,
                              void* d_out, int out_size, void* d_ws, size_t ws_size,
                              hipStream_t stream) {
  const float* x      = (const float*)d_in[0];
  const float* emb_w  = (const float*)d_in[1];
  const float* emb_b  = (const float*)d_in[2];
  const float* cls    = (const float*)d_in[3];
  const float* pos    = (const float*)d_in[4];
  const float* ln1_g  = (const float*)d_in[5];
  const float* ln1_b  = (const float*)d_in[6];
  const float* qkv_w  = (const float*)d_in[7];
  const float* qkv_b  = (const float*)d_in[8];
  const float* out_w  = (const float*)d_in[9];
  const float* out_b  = (const float*)d_in[10];
  const float* ln2_g  = (const float*)d_in[11];
  const float* ln2_b  = (const float*)d_in[12];
  const float* mlp_w1 = (const float*)d_in[13];
  const float* mlp_b1 = (const float*)d_in[14];
  const float* mlp_w2 = (const float*)d_in[15];
  const float* mlp_b2 = (const float*)d_in[16];
  const float* head_w1 = (const float*)d_in[17];
  const float* head_b1 = (const float*)d_in[18];
  const float* head_w2 = (const float*)d_in[19];
  const float* head_b2 = (const float*)d_in[20];
  float* out = (float*)d_out;

  // workspace carve-up (256B aligned)
  char* w = (char*)d_ws;
  auto carve = [&](size_t bytes) -> void* {
    char* p = w;
    w += (bytes + 255) & ~(size_t)255;
    return p;
  };
  _Float16* Xp16  = (_Float16*)carve((size_t)EROWS * DIM * 2);
  float*    Xe    = (float*)carve((size_t)EROWS * DIM * 4);
  float*    h     = (float*)carve((size_t)ROWS * DIM * 4);
  _Float16* a16   = (_Float16*)carve((size_t)ROWS * DIM * 2);
  _Float16* qkv16 = (_Float16*)carve((size_t)ROWS * QKVLD * 2);
  _Float16* o16   = (_Float16*)carve((size_t)ROWS * DIM * 2);
  _Float16* mh16  = (_Float16*)carve((size_t)ROWS * FFDIM * 2);
  _Float16* cls16 = (_Float16*)carve((size_t)BVIT * DIM * 2);
  _Float16* hb16  = (_Float16*)carve((size_t)BVIT * 2 * DIM * 2);

  // 1) patchify + patch embedding
  {
    const size_t n = (size_t)EROWS * DIM;
    patchify_k<<<(unsigned)((n + 255) / 256), 256, 0, stream>>>(x, Xp16);
    gemm_wmma<0, 0><<<ggrid(EROWS, DIM), 256, 0, stream>>>(
        Xp16, emb_w, emb_b, nullptr, Xe, EROWS, DIM, DIM);
    const size_t m = (size_t)ROWS * DIM;
    assemble_k<<<(unsigned)((m + 255) / 256), 256, 0, stream>>>(Xe, cls, pos, h);
  }

  // 2) transformer layers
  const int lnGrid = (ROWS + 7) / 8;
  for (int l = 0; l < NLAYER; ++l) {
    layernorm_rows<<<lnGrid, 256, 0, stream>>>(h, ln1_g + l * DIM, ln1_b + l * DIM,
                                               a16, ROWS);
    gemm_wmma<0, 1><<<ggrid(ROWS, QKVLD), 256, 0, stream>>>(
        a16, qkv_w + (size_t)l * DIM * QKVLD, qkv_b + l * QKVLD, nullptr, qkv16,
        ROWS, QKVLD, DIM);
    attn_wmma<<<BVIT * NHEAD * QTILES, 128, 0, stream>>>(qkv16, o16);
    gemm_wmma<0, 0><<<ggrid(ROWS, DIM), 256, 0, stream>>>(
        o16, out_w + (size_t)l * DIM * DIM, out_b + l * DIM, h, h,
        ROWS, DIM, DIM);
    layernorm_rows<<<lnGrid, 256, 0, stream>>>(h, ln2_g + l * DIM, ln2_b + l * DIM,
                                               a16, ROWS);
    gemm_wmma<1, 1><<<ggrid(ROWS, FFDIM), 256, 0, stream>>>(
        a16, mlp_w1 + (size_t)l * DIM * FFDIM, mlp_b1 + l * FFDIM, nullptr, mh16,
        ROWS, FFDIM, DIM);
    gemm_wmma<0, 0><<<ggrid(ROWS, DIM), 256, 0, stream>>>(
        mh16, mlp_w2 + (size_t)l * FFDIM * DIM, mlp_b2 + l * DIM, h, h,
        ROWS, DIM, FFDIM);
  }

  // 3) head: cls token -> 2D -> NC
  gather_cls_k<<<(BVIT * DIM + 255) / 256, 256, 0, stream>>>(h, cls16);
  gemm_wmma<2, 1><<<ggrid(BVIT, 2 * DIM), 256, 0, stream>>>(
      cls16, head_w1, head_b1, nullptr, hb16, BVIT, 2 * DIM, DIM);
  gemm_wmma<0, 0><<<ggrid(BVIT, NCLS), 256, 0, stream>>>(
      hb16, head_w2, head_b2, nullptr, out, BVIT, NCLS, 2 * DIM);
}
